// CAM_48344151883793
// MI455X (gfx1250) — matrix operand order repaired and checked by reading, not compiled, not run on hardware
//
#include <hip/hip_runtime.h>
#include <hip/hip_bf16.h>

// Problem constants from the reference: B=16, H=W=64 -> N=4096, C=512.
#define BB 16
#define NN 4096
#define CC 512
#define LDP 40   // LDS row pitch in bf16 elems (80B: 16B-aligned, staggers banks)

typedef __bf16 bf16_t;
typedef bf16_t v16bf __attribute__((ext_vector_type(16)));
typedef bf16_t v8bf  __attribute__((ext_vector_type(8)));
typedef bf16_t v2bf  __attribute__((ext_vector_type(2)));
typedef float  v8f   __attribute__((ext_vector_type(8)));

union Frag { v16bf v; uint4 u[2]; };

// Native convert: clang lowers pairs of these to the packed bf16 cvt path.
static __device__ __forceinline__ bf16_t cvt(float x) { return (bf16_t)x; }

// -------------------------------------------------------------------------
// Kernel 1: G[b] = f^T f  (C x C), f = inputs viewed as (B, N, C), fp32 in.
// Block = 256 threads (8 waves) computes a 64(c) x 128(d) tile of G.
// Waves 2x4; each wave computes 32x32 = 4 accumulators -> 4 WMMA / K-step.
// Staging: each thread owns a k-PAIR so transposed writes are packed
// bf16x2 ds_store_b32 (k is the contiguous direction of the [col][k] tiles).
// -------------------------------------------------------------------------
__global__ __launch_bounds__(256) void cam_gram(const float* __restrict__ f,
                                                float* __restrict__ G) {
  const int b  = blockIdx.z;
  const int c0 = blockIdx.y * 64;    // output row tile (c)
  const int d0 = blockIdx.x * 128;   // output col tile (d)
  const float* fb = f + (size_t)b * NN * CC;
  float*       Gb = G + (size_t)b * CC * CC;

  __shared__ __align__(16) bf16_t ldsA[64  * LDP];  // [c][k] transposed
  __shared__ __align__(16) bf16_t ldsB[128 * LDP];  // [d][k] transposed

  const int t    = threadIdx.x;
  const int lane = t & 31;
  const int w    = t >> 5;
  const int wm0  = (w >> 2) * 32;   // 2 rows of waves
  const int wn0  = (w & 3) * 32;    // 4 cols of waves
  const int ml   = lane & 15;
  const int hi   = lane >> 4;

  const int kp = t >> 4;   // 0..15 : k-pair index (k = 2*kp, 2*kp+1)
  const int ch = t & 15;   // 0..15 : column chunk

  v8f acc[2][2] = {};

  for (int k0 = 0; k0 < NN; k0 += 32) {
    __syncthreads();
    {   // Stage A (32k x 64c) and B (32k x 128d), fp32 -> bf16, transposed.
      const float* r0 = fb + (size_t)(k0 + 2 * kp) * CC;
      const float* r1 = r0 + CC;
      float4 a0 = *(const float4*)(r0 + c0 + ch * 4);
      float4 a1 = *(const float4*)(r1 + c0 + ch * 4);
      float4 b0 = *(const float4*)(r0 + d0 + ch * 8);
      float4 b1 = *(const float4*)(r0 + d0 + ch * 8 + 4);
      float4 b2 = *(const float4*)(r1 + d0 + ch * 8);
      float4 b3 = *(const float4*)(r1 + d0 + ch * 8 + 4);
      if (k0 + 32 < NN) {   // prefetch next K tile (global_prefetch_b8)
        __builtin_prefetch(r0 + 32 * CC + c0 + ch * 4, 0, 1);
        __builtin_prefetch(r0 + 32 * CC + d0 + ch * 8, 0, 1);
      }
      float a0v[4] = {a0.x, a0.y, a0.z, a0.w};
      float a1v[4] = {a1.x, a1.y, a1.z, a1.w};
      float b0v[8] = {b0.x, b0.y, b0.z, b0.w, b1.x, b1.y, b1.z, b1.w};
      float b1v[8] = {b2.x, b2.y, b2.z, b2.w, b3.x, b3.y, b3.z, b3.w};
#pragma unroll
      for (int j = 0; j < 4; ++j) {
        v2bf p = {cvt(a0v[j]), cvt(a1v[j])};
        *(v2bf*)&ldsA[(ch * 4 + j) * LDP + 2 * kp] = p;
      }
#pragma unroll
      for (int j = 0; j < 8; ++j) {
        v2bf p = {cvt(b0v[j]), cvt(b1v[j])};
        *(v2bf*)&ldsB[(ch * 8 + j) * LDP + 2 * kp] = p;
      }
    }
    __syncthreads();

    // Fragment gathers use contiguous k order, same for A and B:
    // A lane (m=lane&15, hi=lane>>4): K = hi*16 + {0..15}
    // B lane (n=lane&15, hi):         K = hi*16 + {0..15}
    Frag fa[2], fbg[2];
#pragma unroll
    for (int i = 0; i < 2; ++i) {
      const bf16_t* ap = &ldsA[(wm0 + 16 * i + ml) * LDP];
      fa[i].u[0] = *(const uint4*)(ap + hi * 16);
      fa[i].u[1] = *(const uint4*)(ap + hi * 16 + 8);
      const bf16_t* bp = &ldsB[(wn0 + 16 * i + ml) * LDP + hi * 16];
      fbg[i].u[0] = ((const uint4*)bp)[0];
      fbg[i].u[1] = ((const uint4*)bp)[1];
    }
#pragma unroll
    for (int i = 0; i < 2; ++i)
#pragma unroll
      for (int j = 0; j < 2; ++j)
        acc[i][j] = __builtin_amdgcn_wmma_f32_16x16x32_bf16(
            false, fa[i].v, false, fbg[j].v, (short)0, acc[i][j], false, false);
  }

  // C/D layout: VGPR v, lane L -> row M = v + 8*(L>>4), col N = L&15.
#pragma unroll
  for (int i = 0; i < 2; ++i)
#pragma unroll
    for (int j = 0; j < 2; ++j) {
      const int r = c0 + wm0 + 16 * i + hi * 8;
      const int c = d0 + wn0 + 16 * j + ml;
#pragma unroll
      for (int v = 0; v < 8; ++v)
        Gb[(size_t)(r + v) * CC + c] = acc[i][j][v];
    }
}

// -------------------------------------------------------------------------
// Kernel 2: row softmax of G, emitted as bf16 (B operand of GEMM2).
// One wave per 512-element row; 8 rows per 256-thread block.
// -------------------------------------------------------------------------
__global__ __launch_bounds__(256) void cam_softmax(const float* __restrict__ G,
                                                   bf16_t* __restrict__ attn) {
  const int w    = threadIdx.x >> 5;
  const int lane = threadIdx.x & 31;
  const int row  = blockIdx.x * 8 + w;   // 0 .. B*C-1
  const float* g = G + (size_t)row * CC;

  float v[16];
  float mx = -3.402823466e38f;
#pragma unroll
  for (int i = 0; i < 16; ++i) { v[i] = g[lane + i * 32]; mx = fmaxf(mx, v[i]); }
#pragma unroll
  for (int off = 16; off > 0; off >>= 1) mx = fmaxf(mx, __shfl_xor(mx, off, 32));

  float s = 0.f;
#pragma unroll
  for (int i = 0; i < 16; ++i) { v[i] = __expf(v[i] - mx); s += v[i]; }
#pragma unroll
  for (int off = 16; off > 0; off >>= 1) s += __shfl_xor(s, off, 32);

  const float inv = 1.0f / s;
  bf16_t* o = attn + (size_t)row * CC;
#pragma unroll
  for (int i = 0; i < 16; ++i) o[lane + i * 32] = cvt(v[i] * inv);
}

// -------------------------------------------------------------------------
// Kernel 3: out = gamma * (f @ attn) + inputs.  M=4096, Ncols=512, K=512.
// Block = 256 threads computes a 128(m) x 64(d) tile; waves 4x2, 32x32 each.
// A tile (f) stays row-major in LDS (b128 stores); B tile (attn) transposed.
// -------------------------------------------------------------------------
__global__ __launch_bounds__(256) void cam_out(const float* __restrict__ f,
                                               const bf16_t* __restrict__ attn,
                                               const float* __restrict__ gamma,
                                               float* __restrict__ out) {
  const int b  = blockIdx.z;
  const int m0 = blockIdx.x * 128;  // rows (n-dimension of f)
  const int d0 = blockIdx.y * 64;   // cols (output channel)
  const float*  fb = f    + (size_t)b * NN * CC;
  const bf16_t* ab = attn + (size_t)b * CC * CC;

  __shared__ __align__(16) bf16_t ldsA[128 * LDP];  // [m][k] row-major
  __shared__ __align__(16) bf16_t ldsB[64  * LDP];  // [d][k] transposed

  const int t    = threadIdx.x;
  const int lane = t & 31;
  const int w    = t >> 5;
  const int wm0  = (w >> 1) * 32;   // 4 rows of waves
  const int wn0  = (w & 1) * 32;    // 2 cols of waves
  const int ml   = lane & 15;
  const int hi   = lane >> 4;

  v8f acc[2][2] = {};

  for (int k0 = 0; k0 < CC; k0 += 32) {
    __syncthreads();
    {   // A: 128 rows x 32 k, fp32 -> bf16, row-major (2x 16B ds_store each).
      const int row = t >> 1, ach = t & 1;    // ach selects 16-wide k chunk
      const float4* pa = (const float4*)(fb + (size_t)(m0 + row) * CC + k0 + ach * 16);
      float4 a0 = pa[0], a1 = pa[1], a2 = pa[2], a3 = pa[3];
      if (k0 + 32 < CC)
        __builtin_prefetch((const float*)pa + 32, 0, 1);  // next K tile, same row
      float av[16] = {a0.x, a0.y, a0.z, a0.w, a1.x, a1.y, a1.z, a1.w,
                      a2.x, a2.y, a2.z, a2.w, a3.x, a3.y, a3.z, a3.w};
      v8bf p0, p1;
#pragma unroll
      for (int j = 0; j < 8; ++j) { p0[j] = cvt(av[j]); p1[j] = cvt(av[8 + j]); }
      *(v8bf*)&ldsA[row * LDP + ach * 16]     = p0;
      *(v8bf*)&ldsA[row * LDP + ach * 16 + 8] = p1;
    }
    {   // B: attn bf16, 32 k-rows x 64 d-cols, k-pairs -> packed b32 stores.
      const int kp = t >> 4, bch = t & 15;    // k = 2*kp, 4 cols per thread
      const bf16_t* r0 = ab + (size_t)(k0 + 2 * kp) * CC + d0 + bch * 4;
      uint2 w0 = *(const uint2*)r0;           // 4 bf16 at k
      uint2 w1 = *(const uint2*)(r0 + CC);    // 4 bf16 at k+1
      if (k0 + 32 < CC)
        __builtin_prefetch(r0 + 32 * CC, 0, 1);
      __align__(8) bf16_t t0[4], t1[4];
      *(uint2*)t0 = w0;
      *(uint2*)t1 = w1;
#pragma unroll
      for (int j = 0; j < 4; ++j) {
        v2bf p = {t0[j], t1[j]};
        *(v2bf*)&ldsB[(bch * 4 + j) * LDP + 2 * kp] = p;
      }
    }
    __syncthreads();

    Frag fa[2], fbg[2];
#pragma unroll
    for (int i = 0; i < 2; ++i) {
      const bf16_t* ap = &ldsA[(wm0 + 16 * i + ml) * LDP];
      fa[i].u[0] = *(const uint4*)(ap + hi * 16);
      fa[i].u[1] = *(const uint4*)(ap + hi * 16 + 8);
      const bf16_t* bp = &ldsB[(wn0 + 16 * i + ml) * LDP + hi * 16];
      fbg[i].u[0] = ((const uint4*)bp)[0];
      fbg[i].u[1] = ((const uint4*)bp)[1];
    }
#pragma unroll
    for (int i = 0; i < 2; ++i)
#pragma unroll
      for (int j = 0; j < 2; ++j)
        acc[i][j] = __builtin_amdgcn_wmma_f32_16x16x32_bf16(
            false, fa[i].v, false, fbg[j].v, (short)0, acc[i][j], false, false);
  }

  const float gm = gamma[0];
#pragma unroll
  for (int i = 0; i < 2; ++i)
#pragma unroll
    for (int j = 0; j < 2; ++j) {
      const int r = m0 + wm0 + 16 * i + hi * 8;
      const int c = d0 + wn0 + 16 * j + ml;
#pragma unroll
      for (int v = 0; v < 8; ++v) {
        size_t idx = ((size_t)b * NN + (r + v)) * CC + c;
        out[idx] = gm * acc[i][j][v] + f[idx];
      }
    }
}

// -------------------------------------------------------------------------
// Launcher. d_in[0] = inputs fp32 (B*N*C), d_in[1] = gamma fp32 (1).
// Workspace: G fp32 (16.78 MB) then attn bf16 (8.39 MB) = 25.2 MB total.
// -------------------------------------------------------------------------
extern "C" void kernel_launch(void* const* d_in, const int* in_sizes, int n_in,
                              void* d_out, int out_size, void* d_ws, size_t ws_size,
                              hipStream_t stream) {
  const float* inputs = (const float*)d_in[0];
  const float* gamma  = (const float*)d_in[1];
  float* out = (float*)d_out;

  float*  G    = (float*)d_ws;
  bf16_t* attn = (bf16_t*)((char*)d_ws + (size_t)BB * CC * CC * sizeof(float));

  dim3 gGram(CC / 128, CC / 64, BB);          // 4 x 8 x 16
  cam_gram<<<gGram, 256, 0, stream>>>(inputs, G);

  cam_softmax<<<BB * CC / 8, 256, 0, stream>>>(G, attn);

  dim3 gOut(NN / 128, CC / 64, BB);           // 32 x 8 x 16
  cam_out<<<gOut, 256, 0, stream>>>(inputs, attn, gamma, out);
}
